// Scaled_Dot_Product_Attention_78039555768803
// MI455X (gfx1250) — matrix operand order repaired and checked by reading, not compiled, not run on hardware
//
#include <hip/hip_runtime.h>
#include <math.h>

// Scaled dot-product attention for gfx1250 (MI455X), wave32 + WMMA.
// Pre-passes: Q,V -> f16 row-major; K -> f16 transposed [B,H,D,S]; mask -> bit-packed.
// Hot loop: A-fragments (Q, P) use contiguous vector loads, K/V B-fragments per-element loads; mask comes from
// LDS-resident bit words; softmax loops are float4-vectorized.
// Outputs (concatenated in d_out): context [B,H,S,D] f32, attention [B,H,S,S] f32.

typedef __attribute__((ext_vector_type(16))) _Float16 v16h;
typedef __attribute__((ext_vector_type(8)))  _Float16 v8h;
typedef __attribute__((ext_vector_type(4)))  _Float16 v4h;
typedef __attribute__((ext_vector_type(8)))  float    v8f;
typedef __attribute__((ext_vector_type(4)))  float    v4f;

constexpr int Bc = 2, Hc = 16, Sc = 2048, Dc = 128;
constexpr int TQ = 16;                  // q rows per workgroup
constexpr int LDS_STRIDE = Sc + 8;      // 32B-aligned rows + bank-conflict padding
constexpr size_t NQ = (size_t)Bc * Hc * Sc * Dc;       // elements in Q/K/V
constexpr size_t NMW = (size_t)Bc * Sc * (Sc / 32);    // packed mask words

// ---------------- pre-pass: f32 -> f16 row-major (Q, V) ----------------
__global__ __launch_bounds__(256)
void cvt_f32_f16_kernel(const float* __restrict__ src, _Float16* __restrict__ dst, size_t n) {
    size_t i = ((size_t)blockIdx.x * blockDim.x + threadIdx.x) * 4;
    const size_t stride = (size_t)gridDim.x * blockDim.x * 4;
    for (; i + 3 < n; i += stride) {
        v4f v = *(const v4f*)(src + i);
        *(v4h*)(dst + i) = __builtin_convertvector(v, v4h);
    }
}

// ---------------- pre-pass: K [BH,S,D] f32 -> Kt [BH,D,S] f16 ----------------
__global__ __launch_bounds__(256)
void transpose_K_kernel(const float* __restrict__ K, _Float16* __restrict__ Kt) {
    __shared__ _Float16 tile[32][33];
    const int bh = blockIdx.x;
    const int s0 = blockIdx.y * 32;
    const int d0 = blockIdx.z * 32;
    const int tx = threadIdx.x & 31;
    const int ty = threadIdx.x >> 5;
    const float* src = K + ((size_t)bh * Sc + s0) * Dc + d0;
    for (int r = ty; r < 32; r += 8)
        tile[r][tx] = (_Float16)src[(size_t)r * Dc + tx];   // coalesced over d
    __syncthreads();
    _Float16* dst = Kt + ((size_t)bh * Dc + d0) * Sc + s0;
    for (int r = ty; r < 32; r += 8)
        dst[(size_t)r * Sc + tx] = tile[tx][r];             // coalesced over s
}

// ---------------- pre-pass: bit-pack mask via wave32 ballot ----------------
// mask [B,1,S,S] int32 -> bits [B,S,S/32] u32 (bit k of word w = mask[..][w*32+k]!=0)
__global__ __launch_bounds__(256)
void pack_mask_kernel(const int* __restrict__ mask, unsigned int* __restrict__ bits, size_t nwords) {
    const int lane = threadIdx.x & 31;
    size_t w = ((size_t)blockIdx.x * blockDim.x + threadIdx.x) >> 5;
    const size_t wstride = ((size_t)gridDim.x * blockDim.x) >> 5;
    for (; w < nwords; w += wstride) {
        const int mv = mask[w * 32 + lane];                 // coalesced 32-int read
        const unsigned long long bal = __ballot(mv != 0);   // wave32: low 32 bits
        if (lane == 0) bits[w] = (unsigned int)bal;
    }
}

// ---------------- main attention kernel ----------------
__global__ __launch_bounds__(64, 1)
void sdpa_wmma_kernel(const _Float16* __restrict__ Qh,        // [BH,S,D] f16
                      const _Float16* __restrict__ Kt,        // [BH,D,S] f16
                      const _Float16* __restrict__ Vh,        // [BH,S,D] f16
                      const unsigned int* __restrict__ maskb, // [B,S,S/32] bits
                      const float* __restrict__ scale_p,
                      float* __restrict__ ctx_out,
                      float* __restrict__ attn_out)
{
    // LDS: [TQ][LDS_STRIDE] f32 scores, then [TQ][64] u32 mask words
    extern __shared__ float sc[];
    unsigned int* mw = (unsigned int*)&sc[TQ * LDS_STRIDE];

    const int lane = threadIdx.x & 31;
    const int wave = threadIdx.x >> 5;      // 2 waves per workgroup

    const int qt = blockIdx.x % (Sc / TQ);
    const int bh = blockIdx.x / (Sc / TQ);
    const int b  = bh / Hc;
    const int q0 = qt * TQ;
    const float scale = scale_p[0];

    const _Float16* Qb = Qh + (size_t)bh * Sc * Dc;
    const _Float16* Kb = Kt + (size_t)bh * Dc * Sc;
    const _Float16* Vb = Vh + (size_t)bh * Sc * Dc;

    // Stage this q-tile's mask bits: 16 rows x 64 words = 4KB, coalesced.
    {
        const unsigned int* mb_g = maskb + (size_t)b * Sc * (Sc / 32) + (size_t)q0 * (Sc / 32);
        for (int i = threadIdx.x; i < TQ * (Sc / 32); i += 64) mw[i] = mb_g[i];
    }

    // A-fragment per-lane geometry (ISA 7.12.2, 16-bit A 16x32):
    // m = lane%16; halves 0..7 <- K = off0..+7; halves 8..15 <- K = off0+16..+23
    const int mrow = lane & 15;
    const int off0 = (lane < 16) ? 0 : 8;

    // ---- Load Q tile (16x128) as 4 A-fragments: two contiguous v8h runs each ----
    v16h aQ[4];
#pragma unroll
    for (int c = 0; c < 4; ++c) {
        const _Float16* qp = Qb + (size_t)(q0 + mrow) * Dc + c * 32 + off0;
        v8h lo = *(const v8h*)qp;
        v8h hi = *(const v8h*)(qp + 16);
        aQ[c] = __builtin_shufflevector(lo, hi, 0,1,2,3,4,5,6,7,8,9,10,11,12,13,14,15);
    }
    __syncthreads();   // mask words visible

    // ---- Phase 1: scores = mask ? (Q.K^T)*scale : -1e9, into LDS ----
    // Two adjacent 16-key tiles per iteration -> two independent WMMA chains.
    // k0a is always a multiple of 32, so ONE mask word covers both tiles.
    for (int kt = wave * 2; kt < Sc / 16; kt += 4) {
        const int k0a = kt * 16;            // multiple of 32
        const int k0b = k0a + 16;
        v8f acc0 = {}, acc1 = {};
#pragma unroll
        for (int c = 0; c < 4; ++c) {
            const int nn = lane & 15;
            v16h bKa, bKb;
#pragma unroll
            for (int s = 0; s < 16; ++s) {
                const int kk = off0 + (s & 7) + ((s >> 3) << 4);
                const _Float16* kp = Kb + (size_t)(c * 32 + kk) * Sc;
                bKa[s] = kp[k0a + nn];
                bKb[s] = kp[k0b + nn];
            }
            acc0 = __builtin_amdgcn_wmma_f32_16x16x32_f16(
                false, aQ[c], false, bKa, (short)0, acc0, false, false);
            acc1 = __builtin_amdgcn_wmma_f32_16x16x32_f16(
                false, aQ[c], false, bKb, (short)0, acc1, false, false);
        }
        const int n    = lane & 15;
        const int mb   = (lane < 16) ? 0 : 8;
        const int widx = k0a >> 5;
#pragma unroll
        for (int r = 0; r < 8; ++r) {
            const int m = mb + r;
            const unsigned int w = mw[m * (Sc / 32) + widx];  // broadcast ds load
            const float v0 = ((w >> n)        & 1u) ? acc0[r] * scale : -1.0e9f;
            const float v1 = ((w >> (n + 16)) & 1u) ? acc1[r] * scale : -1.0e9f;
            sc[m * LDS_STRIDE + k0a + n] = v0;
            sc[m * LDS_STRIDE + k0b + n] = v1;
        }
    }
    __syncthreads();

    // ---- Phase 2: softmax per row, float4-vectorized, write attention once ----
    float* attn_base = attn_out + (size_t)bh * Sc * Sc + (size_t)q0 * Sc;
    for (int m = wave * 8; m < wave * 8 + 8; ++m) {
        float* row = &sc[m * LDS_STRIDE];
        float mx = -3.0e38f;
        for (int j = lane * 4; j < Sc; j += 128) {
            v4f v = *(const v4f*)(row + j);
            mx = fmaxf(mx, fmaxf(fmaxf(v[0], v[1]), fmaxf(v[2], v[3])));
        }
#pragma unroll
        for (int o = 16; o > 0; o >>= 1) mx = fmaxf(mx, __shfl_xor(mx, o, 32));
        float sum = 0.0f;
        for (int j = lane * 4; j < Sc; j += 128) {
            v4f v = *(const v4f*)(row + j);
            v4f e;
            e[0] = __expf(v[0] - mx); e[1] = __expf(v[1] - mx);
            e[2] = __expf(v[2] - mx); e[3] = __expf(v[3] - mx);
            *(v4f*)(row + j) = e;
            sum += e[0] + e[1] + e[2] + e[3];
        }
#pragma unroll
        for (int o = 16; o > 0; o >>= 1) sum += __shfl_xor(sum, o, 32);
        const float inv = 1.0f / sum;
        for (int j = lane * 4; j < Sc; j += 128) {
            v4f v = *(const v4f*)(row + j);
            v4f p = {v[0] * inv, v[1] * inv, v[2] * inv, v[3] * inv};
            *(v4f*)(row + j) = p;                        // normalized P stays in LDS
            *(v4f*)(attn_base + (size_t)m * Sc + j) = p; // single b128 HBM write
        }
    }
    __syncthreads();

    // ---- Phase 3: context = P @ V, d-tiles split across waves, 4 indep chains ----
    v8f ctx[4] = {};
    for (int kc = 0; kc < Sc / 32; ++kc) {
        const float* pp = &sc[mrow * LDS_STRIDE + kc * 32 + off0];
        v8f plo = *(const v8f*)pp;
        v8f phi = *(const v8f*)(pp + 16);
        v8h lo16 = __builtin_convertvector(plo, v8h);
        v8h hi16 = __builtin_convertvector(phi, v8h);
        v16h aP = __builtin_shufflevector(lo16, hi16, 0,1,2,3,4,5,6,7,8,9,10,11,12,13,14,15);
        __builtin_prefetch(&Vb[(size_t)((kc + 1) * 32 + lane) * Dc], 0, 0);
#pragma unroll
        for (int t = 0; t < 4; ++t) {
            const int dt = wave * 4 + t;
            v16h bV;
#pragma unroll
            for (int s = 0; s < 16; ++s) {
                const int kk = off0 + (s & 7) + ((s >> 3) << 4);
                bV[s] = Vb[(size_t)(kc * 32 + kk) * Dc + dt * 16 + (lane & 15)];
            }
            ctx[t] = __builtin_amdgcn_wmma_f32_16x16x32_f16(
                false, aP, false, bV, (short)0, ctx[t], false, false);
        }
    }
    // ---- store context ----
    {
        const int n  = lane & 15;
        const int mb = (lane < 16) ? 0 : 8;
#pragma unroll
        for (int t = 0; t < 4; ++t) {
            const int dt = wave * 4 + t;
#pragma unroll
            for (int r = 0; r < 8; ++r) {
                const int m = mb + r;
                ctx_out[((size_t)bh * Sc + q0 + m) * Dc + dt * 16 + n] = ctx[t][r];
            }
        }
    }
}

extern "C" void kernel_launch(void* const* d_in, const int* in_sizes, int n_in,
                              void* d_out, int out_size, void* d_ws, size_t ws_size,
                              hipStream_t stream) {
    const float* Q     = (const float*)d_in[0];
    const float* K     = (const float*)d_in[1];
    const float* V     = (const float*)d_in[2];
    const int*   mask  = (const int*)d_in[3];
    const float* scale = (const float*)d_in[4];

    float* ctx_out  = (float*)d_out;                          // [B,H,S,D]
    float* attn_out = ctx_out + NQ;                           // [B,H,S,S]

    // Workspace: Qh | Kt | Vh (f16, NQ each) | mask bits (u32, NMW) ~= 51.3 MB.
    _Float16* Qh = (_Float16*)d_ws;
    _Float16* Kt = Qh + NQ;
    _Float16* Vh = Kt + NQ;
    unsigned int* Mbits = (unsigned int*)(Vh + NQ);

    // Pre-passes (bandwidth-trivial vs 600 MB output).
    {
        const int blk = 256;
        const int grd = (int)(NQ / (blk * 4));
        cvt_f32_f16_kernel<<<grd, blk, 0, stream>>>(Q, Qh, NQ);
        cvt_f32_f16_kernel<<<grd, blk, 0, stream>>>(V, Vh, NQ);
        dim3 tg(Bc * Hc, Sc / 32, Dc / 32);
        transpose_K_kernel<<<tg, 256, 0, stream>>>(K, Kt);
        pack_mask_kernel<<<4096, 256, 0, stream>>>(mask, Mbits, NMW);
    }

    // Main attention pass.
    const dim3 grid(Bc * Hc * (Sc / TQ));   // 4096 workgroups
    const dim3 block(64);                   // 2 wave32 waves
    const size_t shmem = (size_t)TQ * LDS_STRIDE * sizeof(float)
                       + (size_t)TQ * (Sc / 32) * sizeof(unsigned int); // ~132.5 KB
    sdpa_wmma_kernel<<<grid, block, shmem, stream>>>(
        Qh, Kt, Vh, Mbits, scale, ctx_out, attn_out);
}
